// MoETransformerBlock_17746804867280
// MI455X (gfx1250) — compile-verified, matrix operand order checked by reading, not run on hardware
//
#include <hip/hip_runtime.h>

// ---------------- problem constants ----------------
#define BB 16
#define NN 577
#define EE 1024
#define HH 16
#define HD 64
#define FF_DIM 4096
#define MTOK (BB * NN)   // 9232 tokens
#define NPAD 608         // keys padded to multiple of 32
#define KT_TILES 37      // ceil(577/16)

typedef __bf16 bf16;
typedef __bf16 bf16x16 __attribute__((ext_vector_type(16)));
typedef float  f32x8   __attribute__((ext_vector_type(8)));
typedef unsigned int u32x4 __attribute__((ext_vector_type(4)));

// gfx1250 async global->LDS copy support (ASYNCcnt-tracked, no VGPR staging)
#if defined(__gfx1250__) && __has_builtin(__builtin_amdgcn_global_load_async_to_lds_b128) && \
    __has_builtin(__builtin_amdgcn_s_wait_asynccnt)
#define HAVE_ASYNC 1
#define AS1 __attribute__((address_space(1)))
#define AS3 __attribute__((address_space(3)))
typedef int v4i128 __attribute__((vector_size(16)));   // builtin's pointee type (<4 x i32>)
#else
#define HAVE_ASYNC 0
#endif

// ---------------- WMMA helpers ----------------
// 16-bit A/B fragment (16x32 / 32x16): lane L holds row/col (L&15),
// halves 0..7 = K base + (L>>4)*8, halves 8..15 = same + 16  -> two b128 loads.
__device__ __forceinline__ bf16x16 load_frag(const bf16* rowptr, int kbase, int lane) {
    int ko = kbase + ((lane >> 4) << 3);
    union { u32x4 u[2]; bf16x16 v; } t;
    t.u[0] = *(const u32x4*)(rowptr + ko);
    t.u[1] = *(const u32x4*)(rowptr + ko + 16);
    return t.v;
}

__device__ __forceinline__ f32x8 wmma_bf16(bf16x16 a, bf16x16 b, f32x8 c) {
    return __builtin_amdgcn_wmma_f32_16x16x32_bf16(false, a, false, b, (short)0, c, false, false);
}

__device__ __forceinline__ float gelu_f(float x) {
    return 0.5f * x * (1.0f + erff(x * 0.70710678118654752f));
}

// ---------------- elementwise conversion / packing ----------------
__global__ void cvt_f32_bf16_kernel(const float* __restrict__ in, bf16* __restrict__ out, long n) {
    long i = (long)blockIdx.x * 256 + threadIdx.x;
    if (i < n) out[i] = (bf16)in[i];
}

// ad_up_w [4, E, 64] -> [E, 4*64] bf16 (row-major over K = expert*64 + h)
__global__ void pack_adup_kernel(const float* __restrict__ in, bf16* __restrict__ out) {
    long i = (long)blockIdx.x * 256 + threadIdx.x;
    if (i >= (long)EE * 256) return;
    int o = (int)(i >> 8);
    int c = (int)(i & 255);
    int x = c >> 6, h = c & 63;
    out[i] = (bf16)in[((long)x * EE + o) * 64 + h];
}

// ---------------- LayerNorm + expert gates (+ optional LoRA down) ----------------
__global__ __launch_bounds__(256) void ln_gate_kernel(
    const float* __restrict__ X, const float* __restrict__ G, const float* __restrict__ Bb,
    bf16* __restrict__ OutB,
    const float* __restrict__ gate_w, float* __restrict__ gates_out,
    const float* __restrict__ lora_A, float* __restrict__ h_out)
{
    int m = blockIdx.x, tid = threadIdx.x;
    __shared__ float red[256];
    __shared__ float nrm[EE];
    __shared__ float slog[4];
    const float* x = X + (long)m * EE;

    float s = 0.f;
    for (int j = tid; j < EE; j += 256) s += x[j];
    red[tid] = s; __syncthreads();
    for (int o = 128; o > 0; o >>= 1) { if (tid < o) red[tid] += red[tid + o]; __syncthreads(); }
    float mean = red[0] * (1.f / EE);
    __syncthreads();

    s = 0.f;
    for (int j = tid; j < EE; j += 256) { float d = x[j] - mean; s += d * d; }
    red[tid] = s; __syncthreads();
    for (int o = 128; o > 0; o >>= 1) { if (tid < o) red[tid] += red[tid + o]; __syncthreads(); }
    float rstd = rsqrtf(red[0] * (1.f / EE) + 1e-6f);
    __syncthreads();

    for (int j = tid; j < EE; j += 256) {
        float v = (x[j] - mean) * rstd * G[j] + Bb[j];
        nrm[j] = v;
        OutB[(long)m * EE + j] = (bf16)v;
    }
    __syncthreads();

    if (tid < 4) {                                   // expert gate logits
        const float* w = gate_w + (long)tid * EE;
        float d = 0.f;
        for (int j = 0; j < EE; ++j) d += nrm[j] * w[j];
        slog[tid] = d;
    }
    if (h_out && tid >= 32 && tid < 96) {            // LoRA down: h[x,r]
        int i = tid - 32;
        const float* w = lora_A + (long)i * EE;
        float d = 0.f;
        for (int j = 0; j < EE; ++j) d += nrm[j] * w[j];
        h_out[(long)m * 64 + i] = d;
    }
    __syncthreads();
    if (tid == 0) {                                  // softmax over 4 experts
        float a0 = slog[0], a1 = slog[1], a2 = slog[2], a3 = slog[3];
        float mx = fmaxf(fmaxf(a0, a1), fmaxf(a2, a3));
        float e0 = __expf(a0 - mx), e1 = __expf(a1 - mx), e2 = __expf(a2 - mx), e3 = __expf(a3 - mx);
        float inv = 1.f / (e0 + e1 + e2 + e3);
        float* go = gates_out + (long)m * 4;
        go[0] = e0 * inv; go[1] = e1 * inv; go[2] = e2 * inv; go[3] = e3 * inv;
    }
}

// ---------------- LoRA delta: qkv += sum_x g_x * (h_x @ B_x^T) ----------------
__global__ __launch_bounds__(256) void lora_delta_kernel(
    const float* __restrict__ lg, const float* __restrict__ lh,
    const float* __restrict__ lora_B, float* __restrict__ qkv)
{
    int m = blockIdx.x, tid = threadIdx.x;
    __shared__ float hg[64];
    if (tid < 64) hg[tid] = lh[(long)m * 64 + tid] * lg[(long)m * 4 + (tid >> 4)];
    __syncthreads();
    int o = blockIdx.y * 256 + tid;
    float acc = 0.f;
    for (int i = 0; i < 64; ++i) {
        int x = i >> 4, r = i & 15;
        acc += hg[i] * lora_B[((long)x * (3 * EE) + o) * 16 + r];
    }
    qkv[(long)m * (3 * EE) + o] += acc;
}

// ---------------- split fused QKV into q,k (row-major) and v (transposed, padded) ----------------
__global__ void split_qkv_kernel(const float* __restrict__ qkv,
                                 bf16* __restrict__ Q, bf16* __restrict__ Km, bf16* __restrict__ VT)
{
    long idx = (long)blockIdx.x * 256 + threadIdx.x;
    if (idx >= (long)MTOK * 3 * EE) return;
    int m = (int)(idx / (3 * EE));
    int o = (int)(idx % (3 * EE));
    int part = o >> 10, rem = o & 1023, h = rem >> 6, hd = rem & 63;
    int b = m / NN, n = m % NN, bh = b * HH + h;
    bf16 v = (bf16)qkv[idx];
    if (part == 0)      Q [((long)bh * NN + n) * HD + hd] = v;
    else if (part == 1) Km[((long)bh * NN + n) * HD + hd] = v;
    else                VT[((long)bh * HD + hd) * NPAD + n] = v;
}

__global__ void pad_vt_kernel(bf16* __restrict__ VT) {
    long idx = (long)blockIdx.x * 256 + threadIdx.x;
    long total = (long)BB * HH * HD * (NPAD - NN);
    if (idx >= total) return;
    long bhd = idx / (NPAD - NN);
    int  j   = (int)(idx % (NPAD - NN));
    VT[bhd * NPAD + NN + j] = (bf16)0.f;
}

// ---------------- attention: one wave per (b,h, 16-query tile) ----------------
__global__ __launch_bounds__(32) void attn_kernel(
    const bf16* __restrict__ Q, const bf16* __restrict__ Kmat,
    const bf16* __restrict__ VT, bf16* __restrict__ CTX)
{
    int lane = threadIdx.x;
    int qt = blockIdx.x, bh = blockIdx.y;
    int l15 = lane & 15, lhi = lane >> 4;
    extern __shared__ char smem[];
    float* sc = (float*)smem;                      // 16 x NPAD fp32 scores
    bf16*  pr = (bf16*)(smem + 16 * NPAD * 4);     // 16 x NPAD bf16 probs

    const bf16* qb = Q    + (long)bh * NN * HD;
    const bf16* kb = Kmat + (long)bh * NN * HD;
    const bf16* vb = VT   + (long)bh * HD * NPAD;

    int qrow = qt * 16 + l15; if (qrow > NN - 1) qrow = NN - 1;
    bf16x16 qa0 = load_frag(qb + (long)qrow * HD, 0,  lane);
    bf16x16 qa1 = load_frag(qb + (long)qrow * HD, 32, lane);

    for (int kt = 0; kt < KT_TILES; ++kt) {
        int krow = kt * 16 + l15; if (krow > NN - 1) krow = NN - 1;
        bf16x16 b0 = load_frag(kb + (long)krow * HD, 0,  lane);
        bf16x16 b1 = load_frag(kb + (long)krow * HD, 32, lane);
        f32x8 acc = {0, 0, 0, 0, 0, 0, 0, 0};
        acc = wmma_bf16(qa0, b0, acc);
        acc = wmma_bf16(qa1, b1, acc);
        bool valid = (kt * 16 + l15) < NN;
        #pragma unroll
        for (int i = 0; i < 8; ++i)
            sc[(i + 8 * lhi) * NPAD + kt * 16 + l15] = valid ? acc[i] * 0.125f : -1e30f;
    }
    __syncthreads();

    if (lane < 16) {                               // row softmax -> bf16 probs
        float* srow = sc + lane * NPAD;
        float mx = -1e30f;
        for (int j = 0; j < NN; ++j) mx = fmaxf(mx, srow[j]);
        float sum = 0.f;
        for (int j = 0; j < NN; ++j) { float e = __expf(srow[j] - mx); srow[j] = e; sum += e; }
        float inv = 1.f / sum;
        bf16* prow = pr + lane * NPAD;
        for (int j = 0; j < NN; ++j) prow[j] = (bf16)(srow[j] * inv);
        for (int j = NN; j < NPAD; ++j) prow[j] = (bf16)0.f;
    }
    __syncthreads();

    f32x8 c[4];
    #pragma unroll
    for (int t = 0; t < 4; ++t) c[t] = (f32x8){0, 0, 0, 0, 0, 0, 0, 0};
    for (int ks = 0; ks < NPAD / 32; ++ks) {
        bf16x16 pa = load_frag(pr + (long)l15 * NPAD, ks * 32, lane);
        #pragma unroll
        for (int nt = 0; nt < 4; ++nt) {
            bf16x16 vv = load_frag(vb + (long)(nt * 16 + l15) * NPAD, ks * 32, lane);
            c[nt] = wmma_bf16(pa, vv, c[nt]);
        }
    }
    int b = bh >> 4, hh = bh & 15;
    #pragma unroll
    for (int nt = 0; nt < 4; ++nt)
        #pragma unroll
        for (int i = 0; i < 8; ++i) {
            int qg = qt * 16 + i + 8 * lhi;
            if (qg < NN)
                CTX[((long)(b * NN + qg)) * EE + hh * 64 + nt * 16 + l15] = (bf16)c[nt][i];
        }
}

// ---------------- generic bf16 WMMA GEMM with mode-switched epilogue ----------------
// out[m,n] = A[m,:] . W[n,:]  (A: MxK row-major, W: NxK row-major)
// mode 0: outF = acc + bias
// mode 1: outF = acc + bias + res
// mode 2: outB = bf16(gelu(acc + bias))
// mode 3: outB = bf16(gelu(acc + bias) * ag[m, n/64])          (adapter down)
// mode 4: outF += acc + sum_x ag[m,x]*aupb[x*E + n]            (adapter up)
//
// Block: 256 threads = 8 waves (4 M x 2 N), block tile 128x64, K-step 32.
// On gfx1250 the operand tiles are staged into double-buffered LDS with
// GLOBAL_LOAD_ASYNC_TO_LDS_B128 (ASYNCcnt), rows padded to 80B to spread banks.
#define LDST 40   // LDS tile row stride in bf16 elements (80 bytes)

__global__ __launch_bounds__(256) void gemm_bf16_kernel(
    const bf16* __restrict__ A, const bf16* __restrict__ W,
    int M, int K, int ldo,
    const float* __restrict__ bias, const float* __restrict__ res,
    const float* __restrict__ ag, const float* __restrict__ aupb,
    float* __restrict__ outF, bf16* __restrict__ outB, int mode)
{
    int tid = threadIdx.x, lane = tid & 31, wid = tid >> 5;
    int wm = wid & 3, wn = wid >> 2;
    int l15 = lane & 15, lhi = lane >> 4;
    long rbase = (long)blockIdx.x * 128 + wm * 32;
    int  cbase = blockIdx.y * 64 + wn * 32;

    f32x8 c00 = {0, 0, 0, 0, 0, 0, 0, 0}, c01 = c00, c10 = c00, c11 = c00;

#if HAVE_ASYNC
    __shared__ bf16 At[2][128 * LDST];
    __shared__ bf16 Bt[2][64 * LDST];
    long rblk = (long)blockIdx.x * 128;
    int  cblk = blockIdx.y * 64;

    // per-thread async-copy assignments (16B chunks)
    int arow0 = tid >> 2,        aseg = tid & 3;  // A rows 0..63
    int arow1 = (tid + 256) >> 2;                 // A rows 64..127
    long ga0 = rblk + arow0; if (ga0 > M - 1) ga0 = M - 1;
    long ga1 = rblk + arow1; if (ga1 > M - 1) ga1 = M - 1;
    int brow = tid >> 2;                          // B rows 0..63
    const bf16* gA0 = A + ga0 * K + aseg * 8;
    const bf16* gA1 = A + ga1 * K + aseg * 8;
    const bf16* gB  = W + (long)(cblk + brow) * K + aseg * 8;

    auto issue = [&](int buf, int k) {
        __builtin_amdgcn_global_load_async_to_lds_b128(
            (AS1 v4i128*)(gA0 + k), (AS3 v4i128*)&At[buf][arow0 * LDST + aseg * 8], 0, 0);
        __builtin_amdgcn_global_load_async_to_lds_b128(
            (AS1 v4i128*)(gA1 + k), (AS3 v4i128*)&At[buf][arow1 * LDST + aseg * 8], 0, 0);
        __builtin_amdgcn_global_load_async_to_lds_b128(
            (AS1 v4i128*)(gB + k),  (AS3 v4i128*)&Bt[buf][brow * LDST + aseg * 8], 0, 0);
    };

    issue(0, 0);
    __builtin_amdgcn_s_wait_asynccnt(0);
    __syncthreads();

    int cur = 0;
    for (int k = 0; k < K; k += 32) {
        if (k + 32 < K) issue(cur ^ 1, k + 32);
        bf16x16 a0 = load_frag(&At[cur][(wm * 32 + l15) * LDST], 0, lane);
        bf16x16 a1 = load_frag(&At[cur][(wm * 32 + 16 + l15) * LDST], 0, lane);
        bf16x16 b0 = load_frag(&Bt[cur][(wn * 32 + l15) * LDST], 0, lane);
        bf16x16 b1 = load_frag(&Bt[cur][(wn * 32 + 16 + l15) * LDST], 0, lane);
        c00 = wmma_bf16(a0, b0, c00);
        c01 = wmma_bf16(a0, b1, c01);
        c10 = wmma_bf16(a1, b0, c10);
        c11 = wmma_bf16(a1, b1, c11);
        __builtin_amdgcn_s_wait_asynccnt(0);
        __syncthreads();
        cur ^= 1;
    }
#else
    long r0 = rbase + l15;       if (r0 > M - 1) r0 = M - 1;
    long r1 = rbase + 16 + l15;  if (r1 > M - 1) r1 = M - 1;
    const bf16* a0p = A + r0 * K;
    const bf16* a1p = A + r1 * K;
    const bf16* b0p = W + (long)(cbase + l15) * K;
    const bf16* b1p = W + (long)(cbase + 16 + l15) * K;
    for (int k = 0; k < K; k += 32) {
        bf16x16 a0 = load_frag(a0p, k, lane);
        bf16x16 a1 = load_frag(a1p, k, lane);
        bf16x16 b0 = load_frag(b0p, k, lane);
        bf16x16 b1 = load_frag(b1p, k, lane);
        c00 = wmma_bf16(a0, b0, c00);
        c01 = wmma_bf16(a0, b1, c01);
        c10 = wmma_bf16(a1, b0, c10);
        c11 = wmma_bf16(a1, b1, c11);
    }
#endif

    #pragma unroll
    for (int t = 0; t < 4; ++t) {
        f32x8 cc = (t == 0) ? c00 : (t == 1) ? c01 : (t == 2) ? c10 : c11;
        int ro = (t >> 1) * 16, co = (t & 1) * 16;
        #pragma unroll
        for (int i = 0; i < 8; ++i) {
            long r = rbase + ro + i + 8 * lhi;
            int  cI = cbase + co + l15;
            if (r < M) {
                float v = cc[i];
                long idx = r * (long)ldo + cI;
                if (mode == 0)      outF[idx] = v + bias[cI];
                else if (mode == 1) outF[idx] = v + bias[cI] + res[idx];
                else if (mode == 2) outB[idx] = (bf16)gelu_f(v + bias[cI]);
                else if (mode == 3) outB[idx] = (bf16)(gelu_f(v + bias[cI]) * ag[r * 4 + (cI >> 6)]);
                else {
                    float gb = 0.f;
                    #pragma unroll
                    for (int x = 0; x < 4; ++x) gb += ag[r * 4 + x] * aupb[x * EE + cI];
                    outF[idx] += v + gb;
                }
            }
        }
    }
}

// ---------------- host-side orchestration ----------------
extern "C" void kernel_launch(void* const* d_in, const int* in_sizes, int n_in,
                              void* d_out, int out_size, void* d_ws, size_t ws_size,
                              hipStream_t stream) {
    (void)in_sizes; (void)n_in; (void)out_size; (void)ws_size;
    const float* tokens      = (const float*)d_in[0];
    const float* ln1_g       = (const float*)d_in[1];
    const float* ln1_b       = (const float*)d_in[2];
    const float* qkv_w       = (const float*)d_in[3];
    const float* qkv_b       = (const float*)d_in[4];
    const float* proj_w      = (const float*)d_in[5];
    const float* proj_b      = (const float*)d_in[6];
    const float* lora_gate_w = (const float*)d_in[7];
    const float* lora_A      = (const float*)d_in[8];
    const float* lora_B      = (const float*)d_in[9];
    const float* ln2_g       = (const float*)d_in[10];
    const float* ln2_b       = (const float*)d_in[11];
    const float* fc1_w       = (const float*)d_in[12];
    const float* fc1_b       = (const float*)d_in[13];
    const float* fc2_w       = (const float*)d_in[14];
    const float* fc2_b       = (const float*)d_in[15];
    const float* ad_gate_w   = (const float*)d_in[16];
    const float* ad_down_w   = (const float*)d_in[17];
    const float* ad_down_b   = (const float*)d_in[18];
    const float* ad_up_w     = (const float*)d_in[19];
    const float* ad_up_b     = (const float*)d_in[20];
    float* out = (float*)d_out;

    char* w = (char*)d_ws;
    size_t cur = 0;
    auto carve = [&](size_t bytes) -> char* {
        char* p = w + cur;
        cur += (bytes + 255) & ~(size_t)255;
        return p;
    };
    bf16*  n1    = (bf16*) carve((size_t)MTOK * EE * 2);
    bf16*  qkvw  = (bf16*) carve((size_t)3 * EE * EE * 2);
    bf16*  projw = (bf16*) carve((size_t)EE * EE * 2);
    bf16*  fc1wb = (bf16*) carve((size_t)FF_DIM * EE * 2);
    bf16*  fc2wb = (bf16*) carve((size_t)EE * FF_DIM * 2);
    bf16*  adwb  = (bf16*) carve((size_t)256 * EE * 2);
    bf16*  aupwb = (bf16*) carve((size_t)EE * 256 * 2);
    float* lgg   = (float*)carve((size_t)MTOK * 4 * 4);
    float* lh    = (float*)carve((size_t)MTOK * 64 * 4);
    float* qkv   = (float*)carve((size_t)MTOK * 3 * EE * 4);
    bf16*  Qb    = (bf16*) carve((size_t)BB * HH * NN * HD * 2);
    bf16*  Kb    = (bf16*) carve((size_t)BB * HH * NN * HD * 2);
    bf16*  VT    = (bf16*) carve((size_t)BB * HH * HD * NPAD * 2);
    bf16*  ctx   = (bf16*) carve((size_t)MTOK * EE * 2);
    float* x1    = (float*)carve((size_t)MTOK * EE * 4);
    bf16*  n2    = (bf16*) carve((size_t)MTOK * EE * 2);
    float* ag    = (float*)carve((size_t)MTOK * 4 * 4);
    bf16*  h1    = (bf16*) carve((size_t)MTOK * FF_DIM * 2);
    bf16*  ah    = (bf16*) carve((size_t)MTOK * 256 * 2);

    auto blocks = [](long n) { return (unsigned)((n + 255) / 256); };

    // weight conversions
    cvt_f32_bf16_kernel<<<blocks((long)3 * EE * EE), 256, 0, stream>>>(qkv_w, qkvw, (long)3 * EE * EE);
    cvt_f32_bf16_kernel<<<blocks((long)EE * EE),     256, 0, stream>>>(proj_w, projw, (long)EE * EE);
    cvt_f32_bf16_kernel<<<blocks((long)FF_DIM * EE), 256, 0, stream>>>(fc1_w, fc1wb, (long)FF_DIM * EE);
    cvt_f32_bf16_kernel<<<blocks((long)EE * FF_DIM), 256, 0, stream>>>(fc2_w, fc2wb, (long)EE * FF_DIM);
    cvt_f32_bf16_kernel<<<blocks((long)256 * EE),    256, 0, stream>>>(ad_down_w, adwb, (long)256 * EE);
    pack_adup_kernel<<<blocks((long)EE * 256), 256, 0, stream>>>(ad_up_w, aupwb);

    // LN1 + LoRA gates + LoRA down
    ln_gate_kernel<<<MTOK, 256, 0, stream>>>(tokens, ln1_g, ln1_b, n1, lora_gate_w, lgg, lora_A, lh);

    // QKV GEMM (fp32 out, bias)
    dim3 gq((MTOK + 127) / 128, (3 * EE) / 64);
    gemm_bf16_kernel<<<gq, 256, 0, stream>>>(n1, qkvw, MTOK, EE, 3 * EE,
                                             qkv_b, nullptr, nullptr, nullptr, qkv, nullptr, 0);
    // LoRA delta add
    dim3 gd(MTOK, (3 * EE) / 256);
    lora_delta_kernel<<<gd, 256, 0, stream>>>(lgg, lh, lora_B, qkv);

    // split + transpose V + pad
    split_qkv_kernel<<<blocks((long)MTOK * 3 * EE), 256, 0, stream>>>(qkv, Qb, Kb, VT);
    pad_vt_kernel<<<blocks((long)BB * HH * HD * (NPAD - NN)), 256, 0, stream>>>(VT);

    // attention
    dim3 ga(KT_TILES, BB * HH);
    size_t attn_lds = (size_t)16 * NPAD * 4 + (size_t)16 * NPAD * 2;
    attn_kernel<<<ga, 32, attn_lds, stream>>>(Qb, Kb, VT, ctx);

    // proj + residual -> x1
    dim3 gp((MTOK + 127) / 128, EE / 64);
    gemm_bf16_kernel<<<gp, 256, 0, stream>>>(ctx, projw, MTOK, EE, EE,
                                             proj_b, tokens, nullptr, nullptr, x1, nullptr, 1);

    // LN2 + adapter gates
    ln_gate_kernel<<<MTOK, 256, 0, stream>>>(x1, ln2_g, ln2_b, n2, ad_gate_w, ag, nullptr, nullptr);

    // fc1 -> gelu -> bf16
    dim3 gf1((MTOK + 127) / 128, FF_DIM / 64);
    gemm_bf16_kernel<<<gf1, 256, 0, stream>>>(n2, fc1wb, MTOK, EE, FF_DIM,
                                              fc1_b, nullptr, nullptr, nullptr, nullptr, h1, 2);
    // fc2 + residual -> d_out (full overwrite: replay-safe)
    dim3 gf2((MTOK + 127) / 128, EE / 64);
    gemm_bf16_kernel<<<gf2, 256, 0, stream>>>(h1, fc2wb, MTOK, FF_DIM, EE,
                                              fc2_b, x1, nullptr, nullptr, out, nullptr, 1);
    // adapter down: gated gelu -> bf16
    dim3 gad((MTOK + 127) / 128, 256 / 64);
    gemm_bf16_kernel<<<gad, 256, 0, stream>>>(n2, adwb, MTOK, EE, 256,
                                              ad_down_b, nullptr, ag, nullptr, nullptr, ah, 3);
    // adapter up: accumulate into d_out with gate-weighted bias
    dim3 gau((MTOK + 127) / 128, EE / 64);
    gemm_bf16_kernel<<<gau, 256, 0, stream>>>(ah, aupwb, MTOK, 256, EE,
                                              nullptr, nullptr, ag, ad_up_b, out, nullptr, 4);
}
